// Head_75247827025924
// MI455X (gfx1250) — compile-verified, matrix operand order checked by reading, not run on hardware
//
#include <hip/hip_runtime.h>

// ---------------- problem constants ----------------
#define BATCH 2048
#define TSEQ  192
#define CEMB  384
#define HS    64
#define NW    12            // waves per block (384 threads)

// ---------------- LDS layout (ushort units) ----------------
#define WT_STRIDE 392       // 384 + 8 pad  (conflict-free b128 row reads)
#define QK_STRIDE 72        // 64 + 8 pad
#define V_STRIDE  200       // 192 + 8 pad
#define PS_STRIDE 40        // 32 + 8 pad

#define WT_MAT   (HS * WT_STRIDE)              // 25088 per matrix
#define WT_OFF   0
#define QS_OFF   (3 * WT_MAT)                  // 75264
#define KS_OFF   (QS_OFF + TSEQ * QK_STRIDE)   // +13824
#define VST_OFF  (KS_OFF + TSEQ * QK_STRIDE)   // +13824
#define PS_OFF   (VST_OFF + HS * V_STRIDE)     // +12800
#define SMEM_US  (PS_OFF + NW * 16 * PS_STRIDE)
#define SMEM_BYTES (SMEM_US * 2)               // 246784 B

typedef __attribute__((ext_vector_type(16))) __bf16       v16bf;
typedef __attribute__((ext_vector_type(8)))  float        v8f;
typedef __attribute__((ext_vector_type(4)))  unsigned int u32x4;
typedef __attribute__((ext_vector_type(4)))  float        f32x4;

// 2x fp32 -> packed bf16x2 in ONE VALU op via the CDNA5 pack-convert
// instruction (no clang builtin exposed for gfx1250, so emit it directly).
__device__ __forceinline__ unsigned int pack2(float lo, float hi) {
  unsigned int r;
  asm("v_cvt_pk_bf16_f32 %0, %1, %2" : "=v"(r) : "v"(lo), "v"(hi));
  return r;
}
// fp32 -> bf16 (single value, low 16 bits of a pack)
__device__ __forceinline__ unsigned short f2bf(float f) {
  return (unsigned short)pack2(f, f);
}

// Build a 16x32 bf16 A-fragment from 16 fp32 values (two 8-wide chunks).
__device__ __forceinline__ v16bf pack_frag(f32x4 a0, f32x4 a1, f32x4 b0, f32x4 b1) {
  union { unsigned int u[8]; v16bf v; } t;
  t.u[0] = pack2(a0.x, a0.y); t.u[1] = pack2(a0.z, a0.w);
  t.u[2] = pack2(a1.x, a1.y); t.u[3] = pack2(a1.z, a1.w);
  t.u[4] = pack2(b0.x, b0.y); t.u[5] = pack2(b0.z, b0.w);
  t.u[6] = pack2(b1.x, b1.y); t.u[7] = pack2(b1.z, b1.w);
  return t.v;
}

// Generic A/B fragment load from LDS: row selected by lane%16, K-chunk halves
// by lane/16 (16-bit WMMA operand layout: lanes<16 hold K {c..c+7, c+16..c+23},
// lanes>=16 hold K {c+8..c+15, c+24..c+31}); each half is one ds_load_b128.
__device__ __forceinline__ v16bf load_frag_lds(const unsigned short* base, int row,
                                               int stride, int kbase, int hsel) {
  const unsigned short* p = base + row * stride + kbase + hsel * 8;
  union { u32x4 q[2]; v16bf v; } t;
  t.q[0] = *(const u32x4*)p;
  t.q[1] = *(const u32x4*)(p + 16);
  return t.v;
}

__device__ __forceinline__ v8f bwmma(v16bf a, v16bf b, v8f c) {
  return __builtin_amdgcn_wmma_f32_16x16x32_bf16(false, a, false, b, (short)0, c,
                                                 false, false);
}

__global__ __launch_bounds__(384) void attn_head_kernel(
    const float* __restrict__ x, const float* __restrict__ Wk,
    const float* __restrict__ Wq, const float* __restrict__ Wv,
    float* __restrict__ out) {
  extern __shared__ unsigned short lds[];

  const int b    = blockIdx.x;
  const int tid  = threadIdx.x;
  const int lane = tid & 31;
  const int wv   = tid >> 5;       // 0..11, one 16-row query tile per wave
  const int lr   = lane & 15;
  const int hsel = lane >> 4;

  // ---- Phase 0: stage W{k,q,v} transposed into LDS as bf16 -------------
  {
    const float* Ws[3] = {Wk, Wq, Wv};
    #pragma unroll
    for (int m = 0; m < 3; ++m) {
      const float* W = Ws[m];
      unsigned short* wt = lds + WT_OFF + m * WT_MAT;
      for (int idx = tid; idx < CEMB * HS; idx += 384) {
        int c = idx >> 6, h = idx & 63;            // W is [C][HS] row-major
        wt[h * WT_STRIDE + c] = f2bf(W[idx]);      // store transposed
      }
    }
  }
  __syncthreads();

  // ---- Phase 1: projections k,q,v for this wave's 16 rows --------------
  const int r = wv;
  v8f acc[12];
  #pragma unroll
  for (int t = 0; t < 12; ++t)
    #pragma unroll
    for (int i = 0; i < 8; ++i) acc[t][i] = 0.0f;

  const float* xrow = x + ((size_t)b * TSEQ + r * 16 + lr) * CEMB;
  #pragma unroll 2
  for (int kc = 0; kc < CEMB; kc += 32) {
    __builtin_prefetch(xrow + kc + 64, 0, 1);      // global_prefetch next chunk
    const int c0 = kc + hsel * 8;
    f32x4 xa0 = *(const f32x4*)(xrow + c0);
    f32x4 xa1 = *(const f32x4*)(xrow + c0 + 4);
    f32x4 xb0 = *(const f32x4*)(xrow + c0 + 16);
    f32x4 xb1 = *(const f32x4*)(xrow + c0 + 20);
    v16bf afrag = pack_frag(xa0, xa1, xb0, xb1);
    #pragma unroll
    for (int m = 0; m < 3; ++m) {
      // keep 4 fragments live in distinct registers so the 8 ds_load_b128s
      // issue as a clause and overlap the WMMAs (partial dscnt waits)
      v16bf bfrag[4];
      #pragma unroll
      for (int n = 0; n < 4; ++n)
        bfrag[n] = load_frag_lds(lds + WT_OFF + m * WT_MAT, n * 16 + lr,
                                 WT_STRIDE, kc, hsel);
      #pragma unroll
      for (int n = 0; n < 4; ++n)
        acc[m * 4 + n] = bwmma(afrag, bfrag[n], acc[m * 4 + n]);
    }
  }

  // scatter results: k -> ks[s][h], q -> qs[t][h]; v stored transposed vst[h][s]
  #pragma unroll
  for (int n = 0; n < 4; ++n) {
    const int col = n * 16 + lr;                   // h index
    #pragma unroll
    for (int i = 0; i < 8; ++i) {
      const int row = r * 16 + i + hsel * 8;       // t / s index
      lds[KS_OFF + row * QK_STRIDE + col] = f2bf(acc[0 * 4 + n][i]);
      lds[QS_OFF + row * QK_STRIDE + col] = f2bf(acc[1 * 4 + n][i]);
    }
    union { unsigned int u[4]; u32x4 v; } pv;      // 8 consecutive s values
    #pragma unroll
    for (int j = 0; j < 4; ++j)
      pv.u[j] = pack2(acc[2 * 4 + n][2 * j], acc[2 * 4 + n][2 * j + 1]);
    *(u32x4*)(lds + VST_OFF + col * V_STRIDE + r * 16 + hsel * 8) = pv.v;
  }
  __syncthreads();

  // ---- Phase 2: causal flash attention for this wave's query tile ------
  const v16bf qa0 = load_frag_lds(lds + QS_OFF, r * 16 + lr, QK_STRIDE, 0, hsel);
  const v16bf qa1 = load_frag_lds(lds + QS_OFF, r * 16 + lr, QK_STRIDE, 32, hsel);

  v8f o[4];
  float mrun[8], lrun[8];
  #pragma unroll
  for (int i = 0; i < 8; ++i) { mrun[i] = -1e30f; lrun[i] = 0.0f; }
  #pragma unroll
  for (int n = 0; n < 4; ++n)
    #pragma unroll
    for (int i = 0; i < 8; ++i) o[n][i] = 0.0f;

  unsigned short* psw = lds + PS_OFF + wv * 16 * PS_STRIDE;
  const float scale = 0.125f;                      // 1/sqrt(64)

  for (int j0 = 0; j0 <= r; j0 += 2) {             // process key tiles in pairs
    v8f s0a, s1a;
    #pragma unroll
    for (int i = 0; i < 8; ++i) { s0a[i] = 0.0f; s1a[i] = 0.0f; }

    // all 4 K fragments live at once -> clause of 8 ds_load_b128 + overlap
    v16bf kb[4];
    kb[0] = load_frag_lds(lds + KS_OFF, j0 * 16 + lr,       QK_STRIDE, 0,  hsel);
    kb[1] = load_frag_lds(lds + KS_OFF, j0 * 16 + lr,       QK_STRIDE, 32, hsel);
    kb[2] = load_frag_lds(lds + KS_OFF, (j0 + 1) * 16 + lr, QK_STRIDE, 0,  hsel);
    kb[3] = load_frag_lds(lds + KS_OFF, (j0 + 1) * 16 + lr, QK_STRIDE, 32, hsel);
    s0a = bwmma(qa0, kb[0], s0a);
    s0a = bwmma(qa1, kb[1], s0a);
    s1a = bwmma(qa0, kb[2], s1a);
    s1a = bwmma(qa1, kb[3], s1a);

    // scale + causal mask + row-max across the pair
    float rmax[8];
    #pragma unroll
    for (int i = 0; i < 8; ++i) {
      const int trow = r * 16 + i + hsel * 8;
      const int sc0 = j0 * 16 + lr, sc1 = sc0 + 16;
      float a = s0a[i] * scale; if (sc0 > trow) a = -1e30f;
      float c = s1a[i] * scale; if (sc1 > trow) c = -1e30f;
      s0a[i] = a; s1a[i] = c;
      float v = fmaxf(a, c);
      v = fmaxf(v, __shfl_xor(v, 1, 32));
      v = fmaxf(v, __shfl_xor(v, 2, 32));
      v = fmaxf(v, __shfl_xor(v, 4, 32));
      v = fmaxf(v, __shfl_xor(v, 8, 32));
      rmax[i] = v;
    }

    // online softmax update
    #pragma unroll
    for (int i = 0; i < 8; ++i) {
      const float mn = fmaxf(mrun[i], rmax[i]);
      const float alpha = __expf(mrun[i] - mn);
      mrun[i] = mn;
      const float p0 = __expf(s0a[i] - mn);
      const float p1 = __expf(s1a[i] - mn);
      s0a[i] = p0; s1a[i] = p1;
      float sp = p0 + p1;
      sp += __shfl_xor(sp, 1, 32);
      sp += __shfl_xor(sp, 2, 32);
      sp += __shfl_xor(sp, 4, 32);
      sp += __shfl_xor(sp, 8, 32);
      lrun[i] = lrun[i] * alpha + sp;
      #pragma unroll
      for (int n = 0; n < 4; ++n) o[n][i] *= alpha;
    }

    // D-layout -> A-layout via wave-private LDS slab
    #pragma unroll
    for (int i = 0; i < 8; ++i) {
      const int m = i + hsel * 8;
      psw[m * PS_STRIDE + lr]      = f2bf(s0a[i]);
      psw[m * PS_STRIDE + 16 + lr] = f2bf(s1a[i]);
    }
    asm volatile("s_wait_dscnt 0" ::: "memory");
    v16bf pa = load_frag_lds(psw, lr, PS_STRIDE, 0, hsel);
    v16bf vb[4];
    #pragma unroll
    for (int n = 0; n < 4; ++n)
      vb[n] = load_frag_lds(lds + VST_OFF, n * 16 + lr, V_STRIDE, j0 * 16, hsel);
    #pragma unroll
    for (int n = 0; n < 4; ++n)
      o[n] = bwmma(pa, vb[n], o[n]);
  }

  // ---- normalize and store ---------------------------------------------
  #pragma unroll
  for (int i = 0; i < 8; ++i) {
    const float inv = 1.0f / lrun[i];
    #pragma unroll
    for (int n = 0; n < 4; ++n) o[n][i] *= inv;
  }
  float* orow = out + ((size_t)b * TSEQ + r * 16 + hsel * 8) * HS;
  #pragma unroll
  for (int n = 0; n < 4; ++n) {
    const int h = n * 16 + lr;
    #pragma unroll
    for (int i = 0; i < 8; ++i) orow[(size_t)i * HS + h] = o[n][i];
  }
}

extern "C" void kernel_launch(void* const* d_in, const int* in_sizes, int n_in,
                              void* d_out, int out_size, void* d_ws, size_t ws_size,
                              hipStream_t stream) {
  const float* x  = (const float*)d_in[0];
  const float* Wk = (const float*)d_in[1];
  const float* Wq = (const float*)d_in[2];
  const float* Wv = (const float*)d_in[3];
  float* out = (float*)d_out;

  // Opt in to >64KB dynamic LDS (WGP has 320KB); idempotent & deterministic.
  (void)hipFuncSetAttribute((const void*)attn_head_kernel,
                            hipFuncAttributeMaxDynamicSharedMemorySize,
                            (int)SMEM_BYTES);
  attn_head_kernel<<<dim3(BATCH), dim3(384), SMEM_BYTES, stream>>>(x, Wk, Wq, Wv, out);
}
